// Rank_loss_7438883356893
// MI455X (gfx1250) — matrix operand order repaired and checked by reading, not compiled, not run on hardware
//
#include <hip/hip_runtime.h>
#include <math.h>

// ---------------------------------------------------------------------------
// Sizes from the reference
// ---------------------------------------------------------------------------
#define N_ROWS   4096
#define D_FEA    512
#define N_CLS    751
#define D_SIDE   1024
#define K_INST   8
#define MARGIN1  0.5f
#define THETA    0.1f

typedef __attribute__((ext_vector_type(16))) __bf16 bf16x16;
typedef __attribute__((ext_vector_type(8)))  __bf16 bf16x8;
typedef __attribute__((ext_vector_type(8)))  float  f32x8;

// ---------------------------------------------------------------------------
// CDNA5 async memory->LDS helpers (ASYNCcnt-tracked, no VGPR data path)
// Builtin signature (from hipcc diagnostic): first param is
//   int __attribute__((vector_size(16))) addrspace(1)*  (global source),
// second is the addrspace(3) LDS destination, then imm offset, imm cpol.
// ---------------------------------------------------------------------------
#if defined(__has_builtin)
#if __has_builtin(__builtin_amdgcn_global_load_async_to_lds_b128)
#define HAVE_ASYNC_LDS_BUILTIN 1
#endif
#if __has_builtin(__builtin_amdgcn_s_wait_asynccnt)
#define HAVE_WAIT_ASYNC_BUILTIN 1
#endif
#endif

typedef int v4i_gcc __attribute__((vector_size(4 * sizeof(int))));
typedef __attribute__((address_space(1))) v4i_gcc as1_v4i;
typedef __attribute__((address_space(3))) v4i_gcc as3_v4i;

__device__ inline void async_copy16(void* lds_dst, const void* gsrc) {
#ifdef HAVE_ASYNC_LDS_BUILTIN
    __builtin_amdgcn_global_load_async_to_lds_b128(
        (as1_v4i*)(size_t)gsrc, (as3_v4i*)(size_t)lds_dst,
        /*offset=*/0, /*cpol=*/0);
#else
    asm volatile("global_load_async_to_lds_b128 %0, %1, off"
                 :
                 : "v"((unsigned int)(size_t)lds_dst),
                   "v"((unsigned long long)(size_t)gsrc)
                 : "memory");
#endif
}

__device__ inline void wait_async0() {
#ifdef HAVE_WAIT_ASYNC_BUILTIN
    __builtin_amdgcn_s_wait_asynccnt(0);
#else
    asm volatile("s_wait_asynccnt 0" ::: "memory");
#endif
}

// ---------------------------------------------------------------------------
// Wave (32-lane) reductions
// ---------------------------------------------------------------------------
__device__ inline float wave_sum(float v) {
    #pragma unroll
    for (int o = 16; o > 0; o >>= 1) v += __shfl_xor(v, o, 32);
    return v;
}
__device__ inline float wave_max(float v) {
    #pragma unroll
    for (int o = 16; o > 0; o >>= 1) v = fmaxf(v, __shfl_xor(v, o, 32));
    return v;
}

// ---------------------------------------------------------------------------
// Kernel 0: zero the scalar accumulators
// scal[0]=rank_sum scal[1]=correct scal[2]=logp_sum scal[3]=ss2 scal[4]=ss3 scal[5]=acc
// ---------------------------------------------------------------------------
__global__ void zero_scal_kernel(float* scal) {
    if (threadIdx.x < 8) scal[threadIdx.x] = 0.0f;
}

// ---------------------------------------------------------------------------
// Kernel 1: fp32 -> bf16 (round-to-nearest-even) for the GEMM input
// ---------------------------------------------------------------------------
__global__ void cvt_bf16_kernel(const float* __restrict__ fea,
                                unsigned short* __restrict__ abf) {
    int idx = blockIdx.x * blockDim.x + threadIdx.x;
    int stride = gridDim.x * blockDim.x;
    for (int i = idx; i < N_ROWS * D_FEA; i += stride) {
        unsigned int u = __float_as_uint(fea[i]);
        unsigned int r = u + 0x7fffu + ((u >> 16) & 1u);
        abf[i] = (unsigned short)(r >> 16);
    }
}

// ---------------------------------------------------------------------------
// Kernel 2: per-row squared L2 norm (one wave per row)
// ---------------------------------------------------------------------------
__global__ __launch_bounds__(256) void rowsq_kernel(const float* __restrict__ fea,
                                                    float* __restrict__ sq) {
    int wv = blockIdx.x * 8 + (threadIdx.x >> 5);
    int lane = threadIdx.x & 31;
    if (wv >= N_ROWS) return;
    const float* row = fea + (size_t)wv * D_FEA;
    float s = 0.0f;
    for (int c = lane; c < D_FEA; c += 32) { float v = row[c]; s += v * v; }
    s = wave_sum(s);
    if (lane == 0) sq[wv] = s;
}

// ---------------------------------------------------------------------------
// Kernel 3: WMMA GEMM  dist = sqrt(max(sq_i + sq_j - 2*A.A^T, 1e-12))
//  - 128x128 output tile per workgroup (256 threads = 8 waves)
//  - wave grid 2x4 -> each wave owns 64x32 = 4x2 tiles of 16x16
//  - DOUBLE-BUFFERED LDS panels filled by global_load_async_to_lds_b128
//    (ASYNCcnt), overlapped with v_wmma_f32_16x16x32_bf16 on current panel
// ---------------------------------------------------------------------------
#define LDSS 40   // ushort row stride (80B: keeps 16B alignment, limits conflicts)
#define PANEL (128 * LDSS)

__device__ inline void stage_panels(const unsigned short* __restrict__ Abf,
                                    unsigned short* As, unsigned short* Bs,
                                    int i0, int j0, int k0,
                                    int srow, int scol) {
    const unsigned short* ga = Abf + (size_t)(i0 + srow) * D_FEA + k0 + scol;
    unsigned short* la = As + srow * LDSS + scol;
    async_copy16(la, ga);
    async_copy16(la + 8, ga + 8);
    const unsigned short* gb = Abf + (size_t)(j0 + srow) * D_FEA + k0 + scol;
    unsigned short* lb = Bs + srow * LDSS + scol;
    async_copy16(lb, gb);
    async_copy16(lb + 8, gb + 8);
}

__global__ __launch_bounds__(256) void gemm_dist_kernel(
        const unsigned short* __restrict__ Abf,
        const float* __restrict__ sq,
        float* __restrict__ dist) {
    __shared__ unsigned short As[2 * PANEL];
    __shared__ unsigned short Bs[2 * PANEL];

    const int tid  = threadIdx.x;
    const int lane = tid & 31;
    const int wv   = tid >> 5;
    const int wi   = wv >> 2;    // 0..1  (M direction, 64 rows each)
    const int wj   = wv & 3;     // 0..3  (N direction, 32 cols each)
    const int i0   = blockIdx.y * 128;
    const int j0   = blockIdx.x * 128;
    const int half = lane >> 4;  // lane group 0-15 / 16-31
    const int l16  = lane & 15;

    f32x8 acc[4][2];
    #pragma unroll
    for (int m = 0; m < 4; ++m)
        #pragma unroll
        for (int n = 0; n < 2; ++n)
            #pragma unroll
            for (int v = 0; v < 8; ++v) acc[m][n][v] = 0.0f;

    // staging: each thread copies 2x16B (async b128) into each panel
    const int srow = tid >> 1;          // 0..127
    const int scol = (tid & 1) * 16;    // 0 or 16 (ushorts)

    // prologue: fill buffer 0 for k0 = 0
    stage_panels(Abf, As, Bs, i0, j0, 0, srow, scol);

    for (int it = 0; it < D_FEA / 32; ++it) {
        const int cur = it & 1;
        unsigned short* Ac = As + cur * PANEL;
        unsigned short* Bc = Bs + cur * PANEL;

        // our async copies into buffer `cur` have landed; barrier makes all
        // waves' copies visible AND fences previous iteration's LDS reads
        wait_async0();
        __syncthreads();

        // prefetch next K-panel into the other buffer, overlapped with WMMA
        if (it + 1 < D_FEA / 32)
            stage_panels(Abf, As + (cur ^ 1) * PANEL, Bs + (cur ^ 1) * PANEL,
                         i0, j0, (it + 1) * 32, srow, scol);

        // A fragment (16x32 bf16): lanes 0-15: M=l16, K={0..7,16..23};
        //                          lanes 16-31: same M, K={8..15,24..31}
        bf16x16 afrag[4];
        #pragma unroll
        for (int m = 0; m < 4; ++m) {
            const unsigned short* rp = Ac + (wi * 64 + m * 16 + l16) * LDSS;
            bf16x8 lo = *(const bf16x8*)(rp + half * 8);
            bf16x8 hi = *(const bf16x8*)(rp + 16 + half * 8);
            #pragma unroll
            for (int e = 0; e < 8; ++e) { afrag[m][e] = lo[e]; afrag[m][e + 8] = hi[e]; }
        }
        // B fragment (32x16 bf16): lanes 0-15: N=l16, K={0..7,16..23};
        //                          lanes 16-31: same N, K={8..15,24..31}
        bf16x16 bfrag[2];
        #pragma unroll
        for (int n = 0; n < 2; ++n) {
            const unsigned short* rp = Bc + (wj * 32 + n * 16 + l16) * LDSS;
            bf16x8 lo = *(const bf16x8*)(rp + half * 8);
            bf16x8 hi = *(const bf16x8*)(rp + 16 + half * 8);
            #pragma unroll
            for (int e = 0; e < 8; ++e) { bfrag[n][e] = lo[e]; bfrag[n][e + 8] = hi[e]; }
        }

        #pragma unroll
        for (int m = 0; m < 4; ++m)
            #pragma unroll
            for (int n = 0; n < 2; ++n)
                acc[m][n] = __builtin_amdgcn_wmma_f32_16x16x32_bf16(
                    false, afrag[m], false, bfrag[n],
                    (short)0, acc[m][n], false, false);
    }

    // Epilogue: C/D layout -> lane holds N=l16; VGPR v holds M = v + half*8
    #pragma unroll
    for (int m = 0; m < 4; ++m) {
        #pragma unroll
        for (int n = 0; n < 2; ++n) {
            const int rbase = i0 + wi * 64 + m * 16 + half * 8;
            const int col   = j0 + wj * 32 + n * 16 + l16;
            const float sc  = sq[col];
            #pragma unroll
            for (int v = 0; v < 8; ++v) {
                const int row = rbase + v;
                float d2 = sq[row] + sc - 2.0f * acc[m][n][v];
                dist[(size_t)row * N_ROWS + col] = sqrtf(fmaxf(d2, 1e-12f));
            }
        }
    }
}

// ---------------------------------------------------------------------------
// Kernel 4: per-row rank mining.  One 256-thread block per row.
//  - counts rank of the 8 positive distances (strictly-smaller counters)
//  - finds the 8 smallest negative distances (local top-8 + LDS tournament)
//  - thread 0 evaluates the closed-form loss
// ---------------------------------------------------------------------------
__device__ inline void ins8(float* a, float v) {
    if (v < a[7]) {
        a[7] = v;
        for (int k = 7; k > 0; --k) {
            if (a[k] < a[k - 1]) { float t = a[k]; a[k] = a[k - 1]; a[k - 1] = t; }
            else break;
        }
    }
}

__global__ __launch_bounds__(256) void mine_kernel(const float* __restrict__ dist,
                                                   float* __restrict__ scal) {
    __shared__ float s_dp[8];
    __shared__ int   s_cnt[8];
    __shared__ float s_cand[256 * 8];
    __shared__ float s_c2[32 * 8];

    const int r   = blockIdx.x;
    const int tid = threadIdx.x;
    const float* drow = dist + (size_t)r * N_ROWS;
    const int base = (r >> 3) * K_INST;   // positive block (PK sampling)

    if (tid < 8) { s_dp[tid] = drow[base + tid]; s_cnt[tid] = 0; }
    __syncthreads();

    float dp[8];
    #pragma unroll
    for (int j = 0; j < 8; ++j) dp[j] = s_dp[j];
    int cnt[8] = {0, 0, 0, 0, 0, 0, 0, 0};
    float nv[8];
    #pragma unroll
    for (int k = 0; k < 8; ++k) nv[k] = 3.0e38f;

    for (int c = tid; c < N_ROWS; c += 256) {
        if (c + 256 < N_ROWS) __builtin_prefetch(drow + c + 256, 0, 0);
        float v = drow[c];
        #pragma unroll
        for (int j = 0; j < 8; ++j) cnt[j] += (v < dp[j]) ? 1 : 0;
        bool pos = (c >= base) && (c < base + K_INST);
        if (!pos) ins8(nv, v);
    }
    #pragma unroll
    for (int j = 0; j < 8; ++j) atomicAdd(&s_cnt[j], cnt[j]);
    #pragma unroll
    for (int k = 0; k < 8; ++k) s_cand[tid * 8 + k] = nv[k];
    __syncthreads();

    if (tid < 32) {
        float t8[8];
        #pragma unroll
        for (int k = 0; k < 8; ++k) t8[k] = 3.0e38f;
        for (int e = 0; e < 64; ++e) ins8(t8, s_cand[tid * 64 + e]);
        #pragma unroll
        for (int k = 0; k < 8; ++k) s_c2[tid * 8 + k] = t8[k];
    }
    __syncthreads();

    if (tid == 0) {
        float wn[8];
        #pragma unroll
        for (int k = 0; k < 8; ++k) wn[k] = 3.0e38f;
        for (int e = 0; e < 256; ++e) ins8(wn, s_c2[e]);

        // m = # negatives inside the global top-8 = 8 - (# positives with rank<8)
        int inTop = 0;
        #pragma unroll
        for (int j = 0; j < 8; ++j) inTop += (s_cnt[j] < 8) ? 1 : 0;
        int m = 8 - inTop;

        // positives outside top-8, in descending-distance (== descending rank) order
        float pv[8]; int pc[8];
        #pragma unroll
        for (int j = 0; j < 8; ++j) { pv[j] = s_dp[j]; pc[j] = s_cnt[j]; }
        for (int a = 0; a < 8; ++a)
            for (int b = a + 1; b < 8; ++b)
                if (pv[b] > pv[a]) {
                    float tf = pv[a]; pv[a] = pv[b]; pv[b] = tf;
                    int   ti = pc[a]; pc[a] = pc[b]; pc[b] = ti;
                }
        float wp[8]; int np = 0;
        for (int j = 0; j < 8; ++j)
            if (pc[j] >= 8 && np < 8) wp[np++] = pv[j];

        int mm = (m < np) ? m : np;
        float loss = 0.0f;
        float mwn = wn[0];
        for (int k = 0; k < mm; ++k) {
            float ratio = (mwn - wn[k]) / (mwn + 1e-12f);
            float w = (float)(m - k) * expf(ratio);
            loss += w * (wp[k] - wn[k]) + MARGIN1;
        }
        atomicAdd(&scal[0], loss);
        atomicAdd(&scal[1], (m == 0) ? 1.0f : 0.0f);
    }
}

// ---------------------------------------------------------------------------
// Kernel 5: cross entropy, one wave per row (751 logits)
// ---------------------------------------------------------------------------
__global__ __launch_bounds__(256) void xent_kernel(const float* __restrict__ cls,
                                                   const int* __restrict__ tgt,
                                                   float* __restrict__ scal) {
    int wv = blockIdx.x * 8 + (threadIdx.x >> 5);
    int lane = threadIdx.x & 31;
    if (wv >= N_ROWS) return;
    const float* x = cls + (size_t)wv * N_CLS;
    float mx = -3.0e38f;
    for (int c = lane; c < N_CLS; c += 32) mx = fmaxf(mx, x[c]);
    mx = wave_max(mx);
    float s = 0.0f;
    for (int c = lane; c < N_CLS; c += 32) s += expf(x[c] - mx);
    s = wave_sum(s);
    if (lane == 0) {
        float lp = x[tgt[wv]] - mx - logf(s);
        atomicAdd(&scal[2], lp);
    }
}

// ---------------------------------------------------------------------------
// Kernel 6: side losses -- sum of squared diffs (sqrt applied in finalize)
// ---------------------------------------------------------------------------
__global__ __launch_bounds__(256) void side_kernel(const float* __restrict__ l2,
                                                   const float* __restrict__ l3,
                                                   const float* __restrict__ l4,
                                                   float* __restrict__ scal) {
    size_t idx = (size_t)blockIdx.x * blockDim.x + threadIdx.x;
    size_t stride = (size_t)gridDim.x * blockDim.x;
    float a = 0.0f, b = 0.0f;
    const size_t total = (size_t)N_ROWS * D_SIDE;
    for (size_t i = idx; i < total; i += stride) {
        float v4 = l4[i];
        float d2 = v4 - l2[i];
        float d3 = v4 - l3[i];
        a += d2 * d2;
        b += d3 * d3;
    }
    a = wave_sum(a);
    b = wave_sum(b);
    if ((threadIdx.x & 31) == 0) {
        atomicAdd(&scal[3], a);
        atomicAdd(&scal[4], b);
    }
}

// ---------------------------------------------------------------------------
// Kernel 7: top-1 accuracy of input_fea vs targets (one wave per row)
// ---------------------------------------------------------------------------
__global__ __launch_bounds__(256) void accmax_kernel(const float* __restrict__ fea,
                                                     const int* __restrict__ tgt,
                                                     float* __restrict__ scal) {
    int wv = blockIdx.x * 8 + (threadIdx.x >> 5);
    int lane = threadIdx.x & 31;
    if (wv >= N_ROWS) return;
    const float* row = fea + (size_t)wv * D_FEA;
    float bv = -3.0e38f; int bi = 0;
    for (int c = lane; c < D_FEA; c += 32) {
        float v = row[c];
        if (v > bv) { bv = v; bi = c; }
    }
    #pragma unroll
    for (int o = 16; o > 0; o >>= 1) {
        float ov = __shfl_xor(bv, o, 32);
        int   oi = __shfl_xor(bi, o, 32);
        if (ov > bv || (ov == bv && oi < bi)) { bv = ov; bi = oi; }
    }
    if (lane == 0)
        atomicAdd(&scal[5], (bi == tgt[wv]) ? 1.0f : 0.0f);
}

// ---------------------------------------------------------------------------
// Kernel 8: combine scalars -> (total, prec2)
// ---------------------------------------------------------------------------
__global__ void finalize_kernel(const float* __restrict__ scal,
                                float* __restrict__ out) {
    float rank = scal[0] / (float)N_ROWS;
    float prec = scal[1] / (float)N_ROWS;
    float xent = -scal[2] / (float)N_ROWS;
    float side = sqrtf(scal[3]) + sqrtf(scal[4]);
    float acc  = scal[5] / (float)N_ROWS;
    out[0] = rank + xent + THETA * side;
    out[1] = fmaxf(prec, acc);
}

// ---------------------------------------------------------------------------
// Host launcher
// ---------------------------------------------------------------------------
extern "C" void kernel_launch(void* const* d_in, const int* in_sizes, int n_in,
                              void* d_out, int out_size, void* d_ws, size_t ws_size,
                              hipStream_t stream) {
    const float* cls_fea = (const float*)d_in[0];
    const float* l2_side = (const float*)d_in[1];
    const float* l3_side = (const float*)d_in[2];
    const float* l4_side = (const float*)d_in[3];
    const float* fea     = (const float*)d_in[4];
    const int*   tgt     = (const int*)d_in[5];

    char* ws = (char*)d_ws;
    const size_t OFF_DIST = 0;                                       // 64 MB
    const size_t OFF_ABF  = (size_t)N_ROWS * N_ROWS * sizeof(float); // 4 MB
    const size_t OFF_SQ   = OFF_ABF + (size_t)N_ROWS * D_FEA * sizeof(unsigned short);
    const size_t OFF_SC   = OFF_SQ + (size_t)N_ROWS * sizeof(float);

    float*          dist = (float*)(ws + OFF_DIST);
    unsigned short* abf  = (unsigned short*)(ws + OFF_ABF);
    float*          sqp  = (float*)(ws + OFF_SQ);
    float*          scal = (float*)(ws + OFF_SC);

    zero_scal_kernel<<<1, 32, 0, stream>>>(scal);
    cvt_bf16_kernel<<<2048, 256, 0, stream>>>(fea, abf);
    rowsq_kernel<<<N_ROWS / 8, 256, 0, stream>>>(fea, sqp);
    gemm_dist_kernel<<<dim3(N_ROWS / 128, N_ROWS / 128), 256, 0, stream>>>(abf, sqp, dist);
    mine_kernel<<<N_ROWS, 256, 0, stream>>>(dist, scal);
    xent_kernel<<<N_ROWS / 8, 256, 0, stream>>>(cls_fea, tgt, scal);
    side_kernel<<<1024, 256, 0, stream>>>(l2_side, l3_side, l4_side, scal);
    accmax_kernel<<<N_ROWS / 8, 256, 0, stream>>>(fea, tgt, scal);
    finalize_kernel<<<1, 1, 0, stream>>>(scal, (float*)d_out);
}
